// TransformerBlock_70102456205562
// MI455X (gfx1250) — matrix operand order repaired and checked by reading, not compiled, not run on hardware
//
#include <hip/hip_runtime.h>

// ---------------------------------------------------------------------------
// Transformer block forward for MI455X (gfx1250), bf16 WMMA + fp32 accumulate
// ---------------------------------------------------------------------------

typedef __bf16 bf16_t;
typedef __attribute__((ext_vector_type(8)))  __bf16 v8bf;
typedef __attribute__((ext_vector_type(16))) __bf16 v16bf;
typedef __attribute__((ext_vector_type(8)))  float  v8f;
typedef __attribute__((ext_vector_type(4)))  int    v4i;

// address-space-qualified vector types for the async-to-LDS builtin
typedef __attribute__((address_space(1))) v4i g_v4i;   // global
typedef __attribute__((address_space(3))) v4i l_v4i;   // LDS

constexpr int CB   = 4;       // batch
constexpr int CS   = 2048;    // seq
constexpr int CD   = 1024;    // model dim
constexpr int CH   = 16;      // heads
constexpr int CHD  = 64;      // head dim
constexpr int CHID = 4096;    // ffn hidden
constexpr int CM   = CB * CS; // 8192 tokens

#if __has_builtin(__builtin_amdgcn_global_load_async_to_lds_b128)
#define HAVE_ASYNC_LDS 1
#else
#define HAVE_ASYNC_LDS 0
#endif

__device__ __forceinline__ void wait_async_then_sync() {
#if __has_builtin(__builtin_amdgcn_s_wait_asynccnt)
  __builtin_amdgcn_s_wait_asynccnt(0);
#endif
  __syncthreads();
}

// ---- fragment helpers (CDNA5 WMMA 16x16x32 bf16 layouts, wave32) ----------
__device__ __forceinline__ v16bf cat16(v8bf lo, v8bf hi) {
  v16bf r;
#pragma unroll
  for (int i = 0; i < 8; ++i) { r[i] = lo[i]; r[8 + i] = hi[i]; }
  return r;
}

// A (16xK row-major, this lane's row pointer). Lanes 0-15: K 0..15;
// lanes 16-31: K 16..31.
__device__ __forceinline__ v16bf load_a_frag(const bf16_t* rowptr, int k0, int lane) {
  const bf16_t* p = rowptr + k0 + ((lane < 16) ? 0 : 16);
  v8bf lo = *reinterpret_cast<const v8bf*>(p);
  v8bf hi = *reinterpret_cast<const v8bf*>(p + 8);
  return cat16(lo, hi);
}

// B (KxN, stored NT: row n is K-contiguous; this lane's column-row pointer).
// Lanes 0-15: K 0..15; lanes 16-31: K 16..31.
__device__ __forceinline__ v16bf load_b_frag(const bf16_t* rowptr, int k0, int lane) {
  const bf16_t* p = rowptr + k0 + ((lane < 16) ? 0 : 16);
  v8bf lo = *reinterpret_cast<const v8bf*>(p);
  v8bf hi = *reinterpret_cast<const v8bf*>(p + 8);
  return cat16(lo, hi);
}

// ---- elementwise fp32 -> bf16 cast ----------------------------------------
__global__ __launch_bounds__(256) void cast_f32_bf16(const float* __restrict__ in,
                                                     bf16_t* __restrict__ out, int n) {
  int i = blockIdx.x * 256 + threadIdx.x;
  if (i < n) out[i] = (bf16_t)in[i];
}

// ---- dual LayerNorm: h = LN1(x), g = LN2(x), both bf16 --------------------
__global__ __launch_bounds__(256)
void ln_dual_kernel(const float* __restrict__ x,
                    const float* __restrict__ w1, const float* __restrict__ b1,
                    const float* __restrict__ w2, const float* __restrict__ b2,
                    bf16_t* __restrict__ h, bf16_t* __restrict__ g) {
  const int lane = threadIdx.x & 31;
  const int wv   = threadIdx.x >> 5;
  const int row  = blockIdx.x * 8 + wv;
  const float* xr = x + (size_t)row * CD;

  float s = 0.f, sq = 0.f;
  float vals[32];
#pragma unroll
  for (int i = 0; i < 32; ++i) {
    float v = xr[i * 32 + lane];
    vals[i] = v; s += v; sq += v * v;
  }
#pragma unroll
  for (int m = 16; m >= 1; m >>= 1) {
    s  += __shfl_xor(s, m, 32);
    sq += __shfl_xor(sq, m, 32);
  }
  float mu   = s * (1.f / CD);
  float var  = sq * (1.f / CD) - mu * mu;
  float rstd = rsqrtf(var + 1e-5f);

  bf16_t* hr = h + (size_t)row * CD;
  bf16_t* gr = g + (size_t)row * CD;
#pragma unroll
  for (int i = 0; i < 32; ++i) {
    int c = i * 32 + lane;
    float xh = (vals[i] - mu) * rstd;
    hr[c] = (bf16_t)(xh * w1[c] + b1[c]);
    gr[c] = (bf16_t)(xh * w2[c] + b2[c]);
  }
}

// ---- generic NT GEMM: out[M,N] = A[M,K] @ B[N,K]^T, bf16 in, f32 acc ------
// Block: 8 waves as 2(M) x 4(N). Wave tile: 64x64 = 16 accumulators.
// Per 32-k step: 16 WMMAs vs 16 b128 loads (B frag reused across 4 m-tiles).
enum { OUT_F32 = 0, OUT_BF16 = 1, OUT_BF16_RELU = 2, OUT_F32_ADD2 = 3 };

template <int MODE>
__global__ __launch_bounds__(256)
void gemm_nt_wmma(const bf16_t* __restrict__ A, const bf16_t* __restrict__ Bm,
                  int M, int N, int K,
                  float* __restrict__ outf, bf16_t* __restrict__ outb,
                  const float* __restrict__ add1, const float* __restrict__ add2) {
  const int lane = threadIdx.x & 31;
  const int wv   = threadIdx.x >> 5;   // 0..7
  const int wm   = wv >> 2;            // 0..1
  const int wn   = wv & 3;             // 0..3
  const int mbase = blockIdx.y * 128 + wm * 64;
  const int nbase = blockIdx.x * 256 + wn * 64;
  const int l16 = lane & 15;

  const bf16_t* arow[4];
  const bf16_t* brow[4];
#pragma unroll
  for (int t = 0; t < 4; ++t) arow[t] = A  + (size_t)(mbase + t * 16 + l16) * K;
#pragma unroll
  for (int j = 0; j < 4; ++j) brow[j] = Bm + (size_t)(nbase + j * 16 + l16) * K;

  v8f acc[4][4] = {};

  for (int k0 = 0; k0 < K; k0 += 32) {
    if ((k0 & 255) == 0 && k0 + 256 < K) {
      // warm the cache hierarchy one stripe ahead (global_prefetch_b8)
#pragma unroll
      for (int t = 0; t < 4; ++t) __builtin_prefetch(arow[t] + k0 + 256, 0, 1);
#pragma unroll
      for (int j = 0; j < 4; ++j) __builtin_prefetch(brow[j] + k0 + 256, 0, 1);
    }
    v16bf a[4];
#pragma unroll
    for (int t = 0; t < 4; ++t) a[t] = load_a_frag(arow[t], k0, lane);
#pragma unroll
    for (int j = 0; j < 4; ++j) {
      v16bf b = load_b_frag(brow[j], k0, lane);
#pragma unroll
      for (int t = 0; t < 4; ++t) {
        acc[t][j] = __builtin_amdgcn_wmma_f32_16x16x32_bf16(false, a[t], false, b,
                                                            (short)0, acc[t][j], false, false);
      }
    }
  }

  const int rhi = (lane >> 4) * 8;
#pragma unroll
  for (int t = 0; t < 4; ++t) {
#pragma unroll
    for (int j = 0; j < 4; ++j) {
#pragma unroll
      for (int r = 0; r < 8; ++r) {
        int row = mbase + t * 16 + rhi + r;
        int col = nbase + j * 16 + l16;
        size_t idx = (size_t)row * N + col;
        float v = acc[t][j][r];
        if constexpr (MODE == OUT_F32) {
          outf[idx] = v;
        } else if constexpr (MODE == OUT_F32_ADD2) {
          outf[idx] = v + add1[idx] + add2[idx];
        } else if constexpr (MODE == OUT_BF16) {
          outb[idx] = (bf16_t)v;
        } else {
          outb[idx] = (bf16_t)(v > 0.f ? v : 0.f);
        }
      }
    }
  }
}

// ---- flash attention: per (b, head, 64-row q block), online softmax -------
__global__ __launch_bounds__(128)
void flash_attn_kernel(const bf16_t* __restrict__ Q, const bf16_t* __restrict__ Kb,
                       const bf16_t* __restrict__ V, bf16_t* __restrict__ O) {
  const int lane = threadIdx.x & 31;
  const int wv   = threadIdx.x >> 5;   // 0..3, 16 q-rows each
  const int qb   = blockIdx.x;         // 0..31
  const int head = blockIdx.y;
  const int b    = blockIdx.z;
  const int l16  = lane & 15;
  const int rhi  = (lane >> 4) * 8;

  __shared__ __align__(16) bf16_t Ks[64][64];       // [ktok][d] K tile (async staged)
  __shared__ __align__(16) bf16_t VsT[64][72];      // [d][ktok] transposed, padded
  __shared__ __align__(16) bf16_t Ps[4][16][64];    // per-wave P tile

  const size_t base = ((size_t)b * CS) * CD + head * CHD;

  // Q A-fragments for this lane's row (reused across all k blocks)
  const bf16_t* qrow = Q + base + (size_t)(qb * 64 + wv * 16 + l16) * CD;
  v16bf qf[2];
#pragma unroll
  for (int s = 0; s < 2; ++s) qf[s] = load_a_frag(qrow, s * 32, lane);

  float mrun[8], lrun[8];
#pragma unroll
  for (int r = 0; r < 8; ++r) { mrun[r] = -3.0e38f; lrun[r] = 0.f; }
  v8f oacc[4] = {};

  for (int kb = 0; kb <= qb; ++kb) {
    // stage K tile into LDS (async DMA when available) and V tile transposed
#pragma unroll
    for (int c = 0; c < 4; ++c) {
      int ch   = c * 128 + threadIdx.x;   // 0..511
      int ktok = ch >> 3;                 // 0..63
      int d8   = (ch & 7) * 8;
      const bf16_t* kg = Kb + base + (size_t)(kb * 64 + ktok) * CD + d8;
#if HAVE_ASYNC_LDS
      __builtin_amdgcn_global_load_async_to_lds_b128(
          (g_v4i*)kg, (l_v4i*)&Ks[ktok][d8], 0, 0);
#else
      *reinterpret_cast<v8bf*>(&Ks[ktok][d8]) = *reinterpret_cast<const v8bf*>(kg);
#endif
      v8bf vvv = *reinterpret_cast<const v8bf*>(V + base + (size_t)(kb * 64 + ktok) * CD + d8);
#pragma unroll
      for (int e = 0; e < 8; ++e) VsT[d8 + e][ktok] = vvv[e];
    }
    wait_async_then_sync();

    // S = Q @ K^T (16 q-rows x 64 k-tokens per wave), K frags from LDS
    v8f sacc[4] = {};
#pragma unroll
    for (int s = 0; s < 2; ++s) {
#pragma unroll
      for (int j = 0; j < 4; ++j) {
        const bf16_t* krow = &Ks[j * 16 + l16][0];
        v16bf bfr = load_b_frag(krow, s * 32, lane);
        sacc[j] = __builtin_amdgcn_wmma_f32_16x16x32_bf16(false, qf[s], false, bfr,
                                                          (short)0, sacc[j], false, false);
      }
    }

    // scale, causal mask, online softmax update
#pragma unroll
    for (int r = 0; r < 8; ++r) {
      float rm = -3.0e38f;
#pragma unroll
      for (int j = 0; j < 4; ++j) {
        float v = sacc[j][r] * 0.125f;   // 1/sqrt(64)
        if (kb == qb) {
          int ql = wv * 16 + rhi + r;
          int kl = j * 16 + l16;
          if (kl > ql) v += -1.0e9f;     // additive causal mask (as reference)
        }
        sacc[j][r] = v;
        rm = fmaxf(rm, v);
      }
#pragma unroll
      for (int m = 8; m >= 1; m >>= 1) rm = fmaxf(rm, __shfl_xor(rm, m, 32));
      float mnew  = fmaxf(mrun[r], rm);
      float alpha = __expf(mrun[r] - mnew);
      float sum = 0.f;
#pragma unroll
      for (int j = 0; j < 4; ++j) {
        float p = __expf(sacc[j][r] - mnew);
        sacc[j][r] = p;
        sum += p;
      }
#pragma unroll
      for (int m = 8; m >= 1; m >>= 1) sum += __shfl_xor(sum, m, 32);
      lrun[r] = lrun[r] * alpha + sum;
      mrun[r] = mnew;
#pragma unroll
      for (int j = 0; j < 4; ++j) oacc[j][r] *= alpha;
      // spill P row (C layout) into LDS for A-layout re-read
#pragma unroll
      for (int j = 0; j < 4; ++j) Ps[wv][rhi + r][j * 16 + l16] = (bf16_t)sacc[j][r];
    }

    // O += P @ V   (per-wave LDS ops are in-order: stores above visible here)
#pragma unroll
    for (int s = 0; s < 2; ++s) {
      const bf16_t* prow = &Ps[wv][l16][0];
      v16bf pa = load_a_frag(prow, s * 32, lane);
#pragma unroll
      for (int j = 0; j < 4; ++j) {
        const bf16_t* vcol = &VsT[j * 16 + l16][0];
        int kk0 = s * 32 + ((lane < 16) ? 0 : 16);
        v8bf lo = *reinterpret_cast<const v8bf*>(vcol + kk0);
        v8bf hi = *reinterpret_cast<const v8bf*>(vcol + kk0 + 8);
        v16bf bb = cat16(lo, hi);
        oacc[j] = __builtin_amdgcn_wmma_f32_16x16x32_bf16(false, pa, false, bb,
                                                          (short)0, oacc[j], false, false);
      }
    }
    __syncthreads();   // protect Ks/VsT before next iteration refill
  }

  // normalize and write attention output (bf16, (B,S,DIM) layout)
  const int qtok = qb * 64 + wv * 16 + rhi;
#pragma unroll
  for (int j = 0; j < 4; ++j) {
#pragma unroll
    for (int r = 0; r < 8; ++r) {
      float v = oacc[j][r] / lrun[r];
      O[base + (size_t)(qtok + r) * CD + j * 16 + l16] = (bf16_t)v;
    }
  }
}

// ---------------------------------------------------------------------------
extern "C" void kernel_launch(void* const* d_in, const int* in_sizes, int n_in,
                              void* d_out, int out_size, void* d_ws, size_t ws_size,
                              hipStream_t stream) {
  (void)in_sizes; (void)n_in; (void)out_size; (void)ws_size;
  const float* x    = (const float*)d_in[0];
  // d_in[1] = additive causal mask: reproduced analytically in flash kernel
  const float* wq   = (const float*)d_in[2];
  const float* wk   = (const float*)d_in[3];
  const float* wvp  = (const float*)d_in[4];
  const float* wo   = (const float*)d_in[5];
  const float* w1   = (const float*)d_in[6];
  const float* w2   = (const float*)d_in[7];
  const float* ln1w = (const float*)d_in[8];
  const float* ln1b = (const float*)d_in[9];
  const float* ln2w = (const float*)d_in[10];
  const float* ln2b = (const float*)d_in[11];

  char* ws = (char*)d_ws;
  size_t off = 0;
  auto alloc = [&](size_t bytes) -> void* {
    void* p = ws + off;
    off += (bytes + 255) & ~(size_t)255;
    return p;
  };

  bf16_t* wq_b  = (bf16_t*)alloc((size_t)CD * CD * 2);
  bf16_t* wk_b  = (bf16_t*)alloc((size_t)CD * CD * 2);
  bf16_t* wv_b  = (bf16_t*)alloc((size_t)CD * CD * 2);
  bf16_t* wo_b  = (bf16_t*)alloc((size_t)CD * CD * 2);
  bf16_t* w1_b  = (bf16_t*)alloc((size_t)CHID * CD * 2);
  bf16_t* w2_b  = (bf16_t*)alloc((size_t)CD * CHID * 2);
  bf16_t* hb    = (bf16_t*)alloc((size_t)CM * CD * 2);
  bf16_t* gb    = (bf16_t*)alloc((size_t)CM * CD * 2);
  bf16_t* Qb    = (bf16_t*)alloc((size_t)CM * CD * 2);
  bf16_t* Kb    = (bf16_t*)alloc((size_t)CM * CD * 2);
  bf16_t* Vb    = (bf16_t*)alloc((size_t)CM * CD * 2);
  bf16_t* attnb = (bf16_t*)alloc((size_t)CM * CD * 2);
  float*  oproj = (float*) alloc((size_t)CM * CD * 4);
  bf16_t* ffn1  = (bf16_t*)alloc((size_t)CM * CHID * 2);

  // 1) weights -> bf16
  cast_f32_bf16<<<CD * CD / 256, 256, 0, stream>>>(wq, wq_b, CD * CD);
  cast_f32_bf16<<<CD * CD / 256, 256, 0, stream>>>(wk, wk_b, CD * CD);
  cast_f32_bf16<<<CD * CD / 256, 256, 0, stream>>>(wvp, wv_b, CD * CD);
  cast_f32_bf16<<<CD * CD / 256, 256, 0, stream>>>(wo, wo_b, CD * CD);
  cast_f32_bf16<<<CHID * CD / 256, 256, 0, stream>>>(w1, w1_b, CHID * CD);
  cast_f32_bf16<<<CD * CHID / 256, 256, 0, stream>>>(w2, w2_b, CD * CHID);

  // 2) h = LN1(x), g = LN2(x)
  ln_dual_kernel<<<CM / 8, 256, 0, stream>>>(x, ln1w, ln1b, ln2w, ln2b, hb, gb);

  // 3) Q/K/V projections
  dim3 gqkv(CD / 256, CM / 128);
  gemm_nt_wmma<OUT_BF16><<<gqkv, 256, 0, stream>>>(hb, wq_b, CM, CD, CD, nullptr, Qb, nullptr, nullptr);
  gemm_nt_wmma<OUT_BF16><<<gqkv, 256, 0, stream>>>(hb, wk_b, CM, CD, CD, nullptr, Kb, nullptr, nullptr);
  gemm_nt_wmma<OUT_BF16><<<gqkv, 256, 0, stream>>>(hb, wv_b, CM, CD, CD, nullptr, Vb, nullptr, nullptr);

  // 4) causal flash attention
  flash_attn_kernel<<<dim3(CS / 64, CH, CB), 128, 0, stream>>>(Qb, Kb, Vb, attnb);

  // 5) output projection -> f32 scratch
  gemm_nt_wmma<OUT_F32><<<gqkv, 256, 0, stream>>>(attnb, wo_b, CM, CD, CD, oproj, nullptr, nullptr, nullptr);

  // 6) FFN1 with ReLU -> bf16
  gemm_nt_wmma<OUT_BF16_RELU><<<dim3(CHID / 256, CM / 128), 256, 0, stream>>>(
      gb, w1_b, CM, CHID, CD, nullptr, ffn1, nullptr, nullptr);

  // 7) FFN2 + residual: out = x + oproj + ffn1 @ w2^T
  gemm_nt_wmma<OUT_F32_ADD2><<<gqkv, 256, 0, stream>>>(
      ffn1, w2_b, CM, CD, CHID, (float*)d_out, nullptr, x, oproj);
}
